// Ernie4_5_Attention_72576357367970
// MI455X (gfx1250) — matrix operand order repaired and checked by reading, not compiled, not run on hardware
//
#include <hip/hip_runtime.h>

// ---------------------------------------------------------------------------
// Ernie4.5 attention block on MI455X (gfx1250, wave32)
// bf16 WMMA everywhere + TDM (tensor_load_to_lds) staging for the GEMMs.
// ---------------------------------------------------------------------------
typedef __bf16 bf16_t;
typedef bf16_t v16bf __attribute__((ext_vector_type(16)));
typedef bf16_t v8bf  __attribute__((ext_vector_type(8)));
typedef float  v8f   __attribute__((ext_vector_type(8)));
typedef unsigned int u32x4 __attribute__((ext_vector_type(4)));
typedef int          i32x4 __attribute__((ext_vector_type(4)));
typedef int          i32x8 __attribute__((ext_vector_type(8)));

constexpr int T_      = 2048;
constexpr int HIDDEN_ = 4096;
constexpr int NH_     = 32;
constexpr int NKV_    = 8;
constexpr int HD_     = 128;
constexpr int QKV_    = NH_ * HD_ + 2 * NKV_ * HD_;   // 6144
constexpr float THETA_ = 500000.0f;
constexpr float SCALE_ = 0.08838834764831845f;        // 128^-0.5

// LDS tile row stride in elements: 32 data + 8 pad (80 B rows -> bank-spread,
// produced directly by TDM pad_enable: 4 DWORD pad every 16 DWORDs).
constexpr int LDS_STRIDE = 40;

static __device__ __forceinline__ v8f zero8() {
    v8f z;
#pragma unroll
    for (int i = 0; i < 8; ++i) z[i] = 0.0f;
    return z;
}

// concat two 16-byte bf16 chunks into one WMMA operand register block
static __device__ __forceinline__ v16bf load2x8(const bf16_t* p0, const bf16_t* p1) {
    v8bf a = *(const v8bf*)p0;
    v8bf b = *(const v8bf*)p1;
    return __builtin_shufflevector(a, b, 0, 1, 2, 3, 4, 5, 6, 7,
                                   8, 9, 10, 11, 12, 13, 14, 15);
}

// ---------------------------------------------------------------------------
// Tensor Data Mover: 2D tile (tile_d0 x tile_d1 elements, 2-byte data) from a
// row-major tensor [tensor_d1][tensor_d0] into LDS with 16-byte row padding.
// D# layout per CDNA5 ISA (group0 128b, group1 256b).
// ---------------------------------------------------------------------------
static __device__ __forceinline__ void tdm_load_tile_2d(
    unsigned lds_off, const bf16_t* tile_ptr,
    unsigned tensor_d0, unsigned tensor_d1, unsigned stride_d0,
    unsigned tile_d0, unsigned tile_d1) {
    unsigned long long ga = (unsigned long long)(uintptr_t)tile_ptr;
    u32x4 g0;
    g0[0] = 1u;                                   // count=1 valid user D#
    g0[1] = lds_off;                              // lds_addr (bytes)
    g0[2] = (unsigned)ga;                         // global_addr[31:0]
    g0[3] = ((unsigned)(ga >> 32) & 0x01FFFFFFu)  // global_addr[56:32]
            | 0x80000000u;                        // type = 2 ("image")
    i32x8 g1;
    g1[0] = (int)((1u << 16)                      // data_size = 2 bytes
                  | (1u << 20)                    // pad_enable
                  | (3u << 22)                    // pad_interval: 16 DWORDs
                  | (3u << 25));                  // pad_amount: 4 DWORDs
    g1[1] = (int)((tensor_d0 & 0xFFFFu) << 16);                    // dim0[15:0]
    g1[2] = (int)(((tensor_d0 >> 16) & 0xFFFFu) |
                  ((tensor_d1 & 0xFFFFu) << 16));                  // dim0 hi | dim1 lo
    g1[3] = (int)(((tensor_d1 >> 16) & 0xFFFFu) |
                  ((tile_d0 & 0xFFFFu) << 16));                    // dim1 hi | tile0
    g1[4] = (int)(tile_d1 & 0xFFFFu);                              // tile1 | tile2=0
    g1[5] = (int)stride_d0;                                        // dim0_stride lo
    g1[6] = 0;                                                     // stride hi | d1str
    g1[7] = 0;
    i32x4 z4;
    z4[0] = z4[1] = z4[2] = z4[3] = 0;
#if defined(__clang_major__) && (__clang_major__ >= 23)
    i32x8 z8;
#pragma unroll
    for (int i = 0; i < 8; ++i) z8[i] = 0;
    __builtin_amdgcn_tensor_load_to_lds(g0, g1, z4, z4, z8, 0);
#else
    __builtin_amdgcn_tensor_load_to_lds(g0, g1, z4, z4, 0);
#endif
}

// ---------------------------------------------------------------------------
// f32 -> bf16 conversion (grid-stride)
// ---------------------------------------------------------------------------
__global__ __launch_bounds__(256) void f32_to_bf16_kernel(
    const float* __restrict__ in, bf16_t* __restrict__ out, size_t n) {
    size_t i = (size_t)blockIdx.x * blockDim.x + threadIdx.x;
    size_t stride = (size_t)gridDim.x * blockDim.x;
    for (; i < n; i += stride) out[i] = (bf16_t)in[i];
}

// ---------------------------------------------------------------------------
// f32 [R][C]  ->  bf16 transposed [C][R]   (32x32 LDS tiles, coalesced)
// ---------------------------------------------------------------------------
__global__ __launch_bounds__(256) void transpose_f32_bf16_kernel(
    const float* __restrict__ in, bf16_t* __restrict__ out, int R, int C) {
    __shared__ bf16_t tile[32][33];
    const int r0 = blockIdx.y * 32, c0 = blockIdx.x * 32;
    const int tx = threadIdx.x & 31, ty = threadIdx.x >> 5;   // 8 rows / pass
#pragma unroll
    for (int i = 0; i < 32; i += 8)
        tile[ty + i][tx] = (bf16_t)in[(size_t)(r0 + ty + i) * C + c0 + tx];
    __syncthreads();
#pragma unroll
    for (int i = 0; i < 32; i += 8)
        out[(size_t)(c0 + ty + i) * R + r0 + tx] = tile[tx][ty + i];
}

// ---------------------------------------------------------------------------
// bf16 GEMM, f32 accumulate: C[M,N] = A[M,K] * Bt[N,K]^T   (Bt pre-transposed)
// 128x128 block tile, BK=32, 256 threads = 8 waves, wave tile 32x64.
// Both LDS tiles filled by the Tensor Data Mover (wave 0 issues, TENSORcnt).
// ---------------------------------------------------------------------------
__global__ __launch_bounds__(256) void gemm_bf16_wmma_kernel(
    const bf16_t* __restrict__ A, const bf16_t* __restrict__ Bt,
    float* __restrict__ C, int M, int N, int K) {
    __shared__ alignas(16) bf16_t As[128 * LDS_STRIDE];   // rows of A tile
    __shared__ alignas(16) bf16_t Bs[128 * LDS_STRIDE];   // rows of Bt tile

    const int tid  = threadIdx.x;
    const int lane = tid & 31;
    const int wv   = tid >> 5;          // 0..7
    const int wm   = wv >> 1;           // 0..3  (wave row)
    const int wn   = wv & 1;            // 0..1  (wave col)
    const int hf   = lane >> 4;         // lane half
    const int ln   = lane & 15;

    const int rowBlk = blockIdx.y * 128;
    const int colBlk = blockIdx.x * 128;

    const unsigned ldsA = (unsigned)(uintptr_t)(void*)&As[0];
    const unsigned ldsB = (unsigned)(uintptr_t)(void*)&Bs[0];

    v8f acc[2][4];
#pragma unroll
    for (int i = 0; i < 2; ++i)
#pragma unroll
        for (int j = 0; j < 4; ++j) acc[i][j] = zero8();

    for (int k0 = 0; k0 < K; k0 += 32) {
        if (wv == 0) {   // TDM ignores EXEC: exactly one wave issues the DMAs
            tdm_load_tile_2d(ldsA, &A[(size_t)rowBlk * K + k0],
                             (unsigned)K, (unsigned)M, (unsigned)K, 32u, 128u);
            tdm_load_tile_2d(ldsB, &Bt[(size_t)colBlk * K + k0],
                             (unsigned)K, (unsigned)N, (unsigned)K, 32u, 128u);
            __builtin_amdgcn_s_wait_tensorcnt((short)0);
        }
        __syncthreads();

        // A fragments 16x32: per lane two contiguous 16B chunks (K hf*8+0..7,
        // hf*8+16..23) -> ds_load_b128 x2
        v16bf af[2];
#pragma unroll
        for (int i = 0; i < 2; ++i) {
            const bf16_t* p = &As[(wm * 32 + i * 16 + ln) * LDS_STRIDE + hf * 8];
            af[i] = load2x8(p, p + 16);
        }
        // B fragments 32x16: lane = output column, K hf*8+0..7, hf*8+16..23
        v16bf bfr[4];
#pragma unroll
        for (int j = 0; j < 4; ++j) {
            const bf16_t* p = &Bs[(wn * 64 + j * 16 + ln) * LDS_STRIDE + hf * 8];
            bfr[j] = load2x8(p, p + 16);
        }
#pragma unroll
        for (int i = 0; i < 2; ++i)
#pragma unroll
            for (int j = 0; j < 4; ++j)
                acc[i][j] = __builtin_amdgcn_wmma_f32_16x16x32_bf16(
                    false, af[i], false, bfr[j], (short)0, acc[i][j], false, false);
        __syncthreads();
    }

    // C layout: VGPR r -> row r + 8*half, col = lane&15
#pragma unroll
    for (int i = 0; i < 2; ++i)
#pragma unroll
        for (int j = 0; j < 4; ++j)
#pragma unroll
            for (int r = 0; r < 8; ++r) {
                const int row = rowBlk + wm * 32 + i * 16 + r + hf * 8;
                const int col = colBlk + wn * 64 + j * 16 + ln;
                C[(size_t)row * N + col] = acc[i][j][r];
            }
}

// ---------------------------------------------------------------------------
// RoPE (interleaved) + layout split:
//   Qb  [NH][T][HD]  bf16 (rotated)
//   Kb  [NKV][T][HD] bf16 (rotated)
//   Vtb [NKV][HD][T] bf16 (transposed for PV B-fragments)
// ---------------------------------------------------------------------------
__global__ __launch_bounds__(256) void rope_split_kernel(
    const float* __restrict__ qkv, const int* __restrict__ positions,
    bf16_t* __restrict__ Qb, bf16_t* __restrict__ Kb, bf16_t* __restrict__ Vtb) {
    const int t = blockIdx.x;
    const float pos = (float)positions[t];
    const float* row = qkv + (size_t)t * QKV_;

    for (int idx = threadIdx.x; idx < NH_ * 64; idx += 256) {
        const int hq = idx >> 6, p = idx & 63;
        const float x1 = row[hq * HD_ + 2 * p];
        const float x2 = row[hq * HD_ + 2 * p + 1];
        const float ang = pos * __powf(THETA_, -(float)p * (1.0f / 64.0f));
        float sn, cs;
        __sincosf(ang, &sn, &cs);
        bf16_t* q = Qb + ((size_t)hq * T_ + t) * HD_;
        q[2 * p]     = (bf16_t)(x1 * cs - x2 * sn);
        q[2 * p + 1] = (bf16_t)(x2 * cs + x1 * sn);
    }
    const float* krow = row + NH_ * HD_;
    for (int idx = threadIdx.x; idx < NKV_ * 64; idx += 256) {
        const int hk = idx >> 6, p = idx & 63;
        const float x1 = krow[hk * HD_ + 2 * p];
        const float x2 = krow[hk * HD_ + 2 * p + 1];
        const float ang = pos * __powf(THETA_, -(float)p * (1.0f / 64.0f));
        float sn, cs;
        __sincosf(ang, &sn, &cs);
        bf16_t* k = Kb + ((size_t)hk * T_ + t) * HD_;
        k[2 * p]     = (bf16_t)(x1 * cs - x2 * sn);
        k[2 * p + 1] = (bf16_t)(x2 * cs + x1 * sn);
    }
    const float* vrow = row + NH_ * HD_ + NKV_ * HD_;
    for (int idx = threadIdx.x; idx < NKV_ * HD_; idx += 256) {
        const int hv = idx >> 7, d = idx & 127;
        Vtb[((size_t)hv * HD_ + d) * T_ + t] = (bf16_t)vrow[hv * HD_ + d];
    }
}

// ---------------------------------------------------------------------------
// Flash attention (causal, GQA g=4), bf16 WMMA for QK^T and PV.
// 128 threads = 4 waves; each wave owns a 16-row query tile; key blocks of 32.
// Online softmax stats replicated across each 16-lane half (WMMA C layout).
// ---------------------------------------------------------------------------
__global__ __launch_bounds__(128) void flash_attn_wmma_kernel(
    const bf16_t* __restrict__ Qb, const bf16_t* __restrict__ Kb,
    const bf16_t* __restrict__ Vtb, bf16_t* __restrict__ Ob) {
    __shared__ alignas(16) bf16_t pTile[4][16 * 32];

    const int lane = threadIdx.x & 31;
    const int wv   = threadIdx.x >> 5;
    const int hf   = lane >> 4;
    const int ln   = lane & 15;
    const int head = blockIdx.y;
    const int kv   = head >> 2;                 // g = NH/NKV = 4
    const int rbase = blockIdx.x * 64 + wv * 16;

    const bf16_t* Q  = Qb + ((size_t)head * T_ + rbase) * HD_;
    const bf16_t* Kp = Kb + (size_t)kv * T_ * HD_;
    const bf16_t* Vt = Vtb + (size_t)kv * HD_ * T_;

    // Q fragments: 4 chunks over HD=128 (K=32 each), vectorized
    v16bf qf[4];
    {
        const bf16_t* qrow = Q + (size_t)ln * HD_;
#pragma unroll
        for (int c = 0; c < 4; ++c) {
            const bf16_t* p = qrow + c * 32 + hf * 8;
            qf[c] = load2x8(p, p + 16);
        }
    }

    float mrow[8], lrow[8];
    v8f o[8];
#pragma unroll
    for (int r = 0; r < 8; ++r) { mrow[r] = -3.0e38f; lrow[r] = 0.0f; }
#pragma unroll
    for (int f = 0; f < 8; ++f) o[f] = zero8();

    const int nkb = ((rbase + 15) >> 5) + 1;    // causal key blocks

    for (int b = 0; b < nkb; ++b) {
        const int keyb = b * 32;

        // ---- S = Q K^T (two 16x16 tiles = 32 keys) ----
        v8f s0 = zero8(), s1 = zero8();
#pragma unroll
        for (int c = 0; c < 4; ++c) {
            const bf16_t* k0p = Kp + (size_t)(keyb + ln) * HD_ + c * 32 + hf * 8;
            const bf16_t* k1p = Kp + (size_t)(keyb + 16 + ln) * HD_ + c * 32 + hf * 8;
            v16bf bk0 = load2x8(k0p, k0p + 16);
            v16bf bk1 = load2x8(k1p, k1p + 16);
            s0 = __builtin_amdgcn_wmma_f32_16x16x32_bf16(
                false, qf[c], false, bk0, (short)0, s0, false, false);
            s1 = __builtin_amdgcn_wmma_f32_16x16x32_bf16(
                false, qf[c], false, bk1, (short)0, s1, false, false);
        }

        // ---- online softmax update ----
        float p0[8], p1[8], alpha[8];
#pragma unroll
        for (int r = 0; r < 8; ++r) {
            const int qrow = rbase + r + hf * 8;
            float v0 = s0[r] * SCALE_;
            float v1 = s1[r] * SCALE_;
            if (keyb + ln > qrow)      v0 = -3.0e38f;
            if (keyb + 16 + ln > qrow) v1 = -3.0e38f;
            float mx = fmaxf(v0, v1);
#pragma unroll
            for (int m = 1; m < 16; m <<= 1) mx = fmaxf(mx, __shfl_xor(mx, m, 32));
            const float mnew = fmaxf(mrow[r], mx);
            const float e0 = __expf(v0 - mnew);
            const float e1 = __expf(v1 - mnew);
            float rs = e0 + e1;
#pragma unroll
            for (int m = 1; m < 16; m <<= 1) rs += __shfl_xor(rs, m, 32);
            alpha[r] = __expf(mrow[r] - mnew);
            lrow[r]  = lrow[r] * alpha[r] + rs;
            mrow[r]  = mnew;
            p0[r] = e0;
            p1[r] = e1;
        }
#pragma unroll
        for (int f = 0; f < 8; ++f)
#pragma unroll
            for (int r = 0; r < 8; ++r) o[f][r] *= alpha[r];

        // ---- P: C-layout -> A-layout via per-wave LDS bounce ----
        bf16_t* pl = &pTile[wv][0];
#pragma unroll
        for (int r = 0; r < 8; ++r) {
            pl[(r + hf * 8) * 32 + ln]      = (bf16_t)p0[r];
            pl[(r + hf * 8) * 32 + 16 + ln] = (bf16_t)p1[r];
        }
        asm volatile("s_wait_dscnt 0" ::: "memory");   // wave-local LDS RAW
        const bf16_t* pp = pl + ln * 32 + hf * 8;
        v16bf pa = load2x8(pp, pp + 16);

        // ---- O += P V  (V^T layout -> contiguous B-fragments) ----
#pragma unroll
        for (int f = 0; f < 8; ++f) {
            const bf16_t* vp = Vt + (size_t)(f * 16 + ln) * T_ + keyb + hf * 8;
            v16bf bv = load2x8(vp, vp + 16);
            o[f] = __builtin_amdgcn_wmma_f32_16x16x32_bf16(
                false, pa, false, bv, (short)0, o[f], false, false);
        }
    }

    // ---- normalize + store bf16 [T][NH*HD] ----
#pragma unroll
    for (int f = 0; f < 8; ++f)
#pragma unroll
        for (int r = 0; r < 8; ++r) {
            const int t   = rbase + r + hf * 8;
            const int col = head * HD_ + f * 16 + ln;
            Ob[(size_t)t * (NH_ * HD_) + col] = (bf16_t)(o[f][r] / lrow[r]);
        }
}

// ---------------------------------------------------------------------------
// Host-side launch
// ---------------------------------------------------------------------------
extern "C" void kernel_launch(void* const* d_in, const int* in_sizes, int n_in,
                              void* d_out, int out_size, void* d_ws, size_t ws_size,
                              hipStream_t stream) {
    const int*   positions = (const int*)d_in[0];
    const float* hidden    = (const float*)d_in[1];
    const float* w_qkv     = (const float*)d_in[2];
    const float* w_o       = (const float*)d_in[3];
    float*       out       = (float*)d_out;

    char* ws = (char*)d_ws;
    size_t off = 0;
    auto carve = [&](size_t bytes) {
        char* p = ws + off;
        off += (bytes + 255) & ~(size_t)255;
        return p;
    };
    bf16_t* hb     = (bf16_t*)carve((size_t)T_ * HIDDEN_ * 2);
    bf16_t* wqkvT  = (bf16_t*)carve((size_t)QKV_ * HIDDEN_ * 2);    // [QKV][HIDDEN]
    bf16_t* woT    = (bf16_t*)carve((size_t)HIDDEN_ * HIDDEN_ * 2); // [HIDDEN][4096]
    float*  qkvf   = (float*)carve((size_t)T_ * QKV_ * 4);
    bf16_t* qb     = (bf16_t*)carve((size_t)NH_ * T_ * HD_ * 2);
    bf16_t* kb     = (bf16_t*)carve((size_t)NKV_ * T_ * HD_ * 2);
    bf16_t* vtb    = (bf16_t*)carve((size_t)NKV_ * HD_ * T_ * 2);
    bf16_t* attnb  = (bf16_t*)carve((size_t)T_ * NH_ * HD_ * 2);
    (void)ws_size; (void)n_in; (void)in_sizes; (void)out_size;

    // 1) precision conversion (+ weight transposes for contiguous B-fragments)
    f32_to_bf16_kernel<<<2048, 256, 0, stream>>>(hidden, hb, (size_t)T_ * HIDDEN_);
    transpose_f32_bf16_kernel<<<dim3(QKV_ / 32, HIDDEN_ / 32), 256, 0, stream>>>(
        w_qkv, wqkvT, HIDDEN_, QKV_);
    transpose_f32_bf16_kernel<<<dim3(HIDDEN_ / 32, HIDDEN_ / 32), 256, 0, stream>>>(
        w_o, woT, HIDDEN_, HIDDEN_);

    // 2) QKV projection: [T,HIDDEN] x [HIDDEN,QKV] -> f32
    gemm_bf16_wmma_kernel<<<dim3(QKV_ / 128, T_ / 128), 256, 0, stream>>>(
        hb, wqkvT, qkvf, T_, QKV_, HIDDEN_);

    // 3) RoPE + head-major split (+ V transpose)
    rope_split_kernel<<<T_, 256, 0, stream>>>(qkvf, positions, qb, kb, vtb);

    // 4) causal GQA flash attention
    flash_attn_wmma_kernel<<<dim3(T_ / 64, NH_), 128, 0, stream>>>(qb, kb, vtb, attnb);

    // 5) O projection: [T,4096] x [4096,HIDDEN] -> f32 out
    gemm_bf16_wmma_kernel<<<dim3(HIDDEN_ / 128, T_ / 128), 256, 0, stream>>>(
        attnb, woT, out, T_, HIDDEN_, NH_ * HD_);
}
